// GrupedQueryAttention_35510789603713
// MI455X (gfx1250) — matrix operand order repaired and checked by reading, not compiled, not run on hardware
//
#include <hip/hip_runtime.h>
#include <hip/hip_bf16.h>

// ---------------- problem constants ----------------
#define BATCH   2
#define SEQ     1024
#define EMBED   4096
#define HEADS   32
#define KVHEADS 8
#define HD      128
#define GROUPS  (HEADS / KVHEADS)
#define BS      (BATCH * SEQ)          // 2048 rows total

typedef __attribute__((ext_vector_type(16))) __bf16          v16bf;
typedef __attribute__((ext_vector_type(8)))  float           v8f;
typedef __attribute__((ext_vector_type(8)))  unsigned short  v8us;
typedef __attribute__((ext_vector_type(16))) unsigned short  v16us;

// f32 -> bf16 round-to-nearest-even (bit trick)
__device__ __forceinline__ unsigned short f32_to_bf16(float f) {
    unsigned int u = __float_as_uint(f);
    u += 0x7FFFu + ((u >> 16) & 1u);
    return (unsigned short)(u >> 16);
}

// Build a 16x bf16 fragment from two 16B-aligned 8-element chunks.
__device__ __forceinline__ v16bf load_frag2(const unsigned short* p0,
                                            const unsigned short* p1) {
    v8us a = *(const v8us*)p0;
    v8us b = *(const v8us*)p1;
    v16us t;
#pragma unroll
    for (int i = 0; i < 8; ++i) { t[i] = a[i]; t[8 + i] = b[i]; }
    return __builtin_bit_cast(v16bf, t);
}

// ---------------- f32 -> bf16 conversion ----------------
__global__ void cvt_bf16_kernel(const float* __restrict__ in,
                                unsigned short* __restrict__ out, size_t n) {
    size_t i = (size_t)blockIdx.x * blockDim.x + threadIdx.x;
    size_t stride = (size_t)gridDim.x * blockDim.x;
    for (; i < n; i += stride) out[i] = f32_to_bf16(in[i]);
}

// ---------------- RoPE: f32 in, bf16 out (optional extra scale) -------------
__global__ void rope_kernel(const float* __restrict__ in,
                            unsigned short* __restrict__ out, int nheads,
                            float out_scale) {
    int t = blockIdx.x * blockDim.x + threadIdx.x;
    int total = BATCH * SEQ * nheads * (HD / 2);
    if (t >= total) return;
    int d  = t & 63;
    int hh = (t >> 6) % nheads;
    int s  = (t / (64 * nheads)) % SEQ;
    int b  = t / (64 * nheads * SEQ);
    size_t base = (((size_t)b * SEQ + s) * nheads + hh) * HD;
    float x1 = in[base + d];
    float x2 = in[base + d + 64];
    float invf = __powf(10000.0f, -((float)(2 * d)) / (float)HD);
    float ang = (float)s * invf;
    float c = __cosf(ang), sn = __sinf(ang);
    out[base + d]      = f32_to_bf16((x1 * c - x2 * sn) * out_scale);
    out[base + d + 64] = f32_to_bf16((x2 * c + x1 * sn) * out_scale);
}

// -------- V (b,s,hk,d) -> V^T (b,hk,d,s), bf16 (one 4 MiB pass) ------------
__global__ void transpose_v_kernel(const unsigned short* __restrict__ in,
                                   unsigned short* __restrict__ out) {
    int t = blockIdx.x * blockDim.x + threadIdx.x;
    if (t >= BATCH * SEQ * KVHEADS * HD) return;
    int d  = t & (HD - 1);
    int hk = (t >> 7) & (KVHEADS - 1);
    int s  = (t >> 10) & (SEQ - 1);
    int b  = t >> 20;
    out[(((size_t)b * KVHEADS + hk) * HD + d) * SEQ + s] = in[t];
}

// ---------------- bf16 WMMA GEMM: C[M,N] = A[M,K] * B[K,N] ----------------
// Block = 256 threads (8 waves). BM=128 x BN=64, BK=64 (2 WMMA K-steps).
// Ping-pong LDS with register staging: one barrier per K-tile.
#define BM   128
#define BN   64
#define BK   64
#define LSTR 72   // padded LDS stride: 144B rows, 16B aligned, bank-friendly

__global__ __launch_bounds__(256)
void gemm_bf16_wmma(const unsigned short* __restrict__ A,
                    const unsigned short* __restrict__ Bm,
                    void* __restrict__ C,
                    int M, int N, int K, int ldc, int out_bf16) {
    __shared__ unsigned short lA[2][BM * LSTR];   // [m][k]
    __shared__ unsigned short lB[2][BN * LSTR];   // [n][k] (transposed B)

    const int tid  = threadIdx.x;
    const int lane = tid & 31;
    const int wave = tid >> 5;
    const int m0 = blockIdx.y * BM;
    const int n0 = blockIdx.x * BN;

    // --- per-thread load geometry, hoisted out of the K loop ---
    // A: 4 x 16B chunks/thread; row = tid>>3 (+32i), col = (tid&7)*8
    const int ar  = tid >> 3;
    const int ac  = (tid & 7) * 8;
    const unsigned short* pA[4];
    int aslot[4];
#pragma unroll
    for (int i = 0; i < 4; ++i) {
        int row = ar + 32 * i;
        pA[i]    = A + (size_t)(m0 + row) * K + ac;
        aslot[i] = row * LSTR + ac;
    }
    // B: 2 x 16B chunks/thread (8 n-values each), scattered into [n][k]
    const int bk_ = tid >> 3;         // k base (+32i)
    const int bn8 = (tid & 7) * 8;    // n base
    const unsigned short* pB[2];
#pragma unroll
    for (int i = 0; i < 2; ++i)
        pB[i] = Bm + (size_t)(bk_ + 32 * i) * N + n0 + bn8;

    v8f acc[4] = {};

    // --- preload tile 0 into buffer 0 ---
    v8us ra[4], rb[2];
#pragma unroll
    for (int i = 0; i < 4; ++i) ra[i] = *(const v8us*)pA[i];
#pragma unroll
    for (int i = 0; i < 2; ++i) rb[i] = *(const v8us*)pB[i];
#pragma unroll
    for (int i = 0; i < 4; ++i) *(v8us*)(&lA[0][aslot[i]]) = ra[i];
#pragma unroll
    for (int i = 0; i < 2; ++i) {
        int kk = bk_ + 32 * i;
#pragma unroll
        for (int j = 0; j < 8; ++j) lB[0][(bn8 + j) * LSTR + kk] = rb[i][j];
    }
    __syncthreads();

    const int T     = K / BK;
    const int mrow  = wave * 16 + (lane & 15);
    const int khalf = (lane >> 4) * 8;
    const int kh2   = (lane >> 4) * 8;
    const int nn    = lane & 15;

    for (int t = 0; t < T; ++t) {
        const int cur = t & 1, nxt = cur ^ 1;
        // stage next tile's global loads (overlaps with WMMA below)
        if (t + 1 < T) {
#pragma unroll
            for (int i = 0; i < 4; ++i)
                ra[i] = *(const v8us*)(pA[i] + (size_t)(t + 1) * BK);
#pragma unroll
            for (int i = 0; i < 2; ++i)
                rb[i] = *(const v8us*)(pB[i] + (size_t)(t + 1) * BK * N);
        }
        // load all fragments, then 8 independent WMMAs
        const unsigned short* la = lA[cur];
        const unsigned short* lb = lB[cur];
        v16bf af[2], bf[2][4];
#pragma unroll
        for (int ks = 0; ks < 2; ++ks) {
            const unsigned short* pa = la + mrow * LSTR + ks * 32 + khalf;
            af[ks] = load_frag2(pa, pa + 16);
#pragma unroll
            for (int j = 0; j < 4; ++j) {
                const unsigned short* pb = lb + (j * 16 + nn) * LSTR + ks * 32 + kh2;
                bf[ks][j] = load_frag2(pb, pb + 16);
            }
        }
#pragma unroll
        for (int ks = 0; ks < 2; ++ks)
#pragma unroll
            for (int j = 0; j < 4; ++j)
                acc[j] = __builtin_amdgcn_wmma_f32_16x16x32_bf16(
                    false, af[ks], false, bf[ks][j], (short)0, acc[j], false, false);
        // commit staged tile into the other buffer
        if (t + 1 < T) {
#pragma unroll
            for (int i = 0; i < 4; ++i) *(v8us*)(&lA[nxt][aslot[i]]) = ra[i];
#pragma unroll
            for (int i = 0; i < 2; ++i) {
                int kk = bk_ + 32 * i;
#pragma unroll
                for (int j = 0; j < 8; ++j) lB[nxt][(bn8 + j) * LSTR + kk] = rb[i][j];
            }
        }
        __syncthreads();
    }

    // C layout: VGPR r -> row r (lanes 0-15) / r+8 (lanes 16-31), col = lane&15
    const int mbase = m0 + wave * 16 + ((lane >> 4) * 8);
    if (out_bf16) {
        unsigned short* Cb = (unsigned short*)C;
#pragma unroll
        for (int j = 0; j < 4; ++j)
#pragma unroll
            for (int r = 0; r < 8; ++r)
                Cb[(size_t)(mbase + r) * ldc + n0 + j * 16 + nn] =
                    f32_to_bf16(acc[j][r]);
    } else {
        float* Cf = (float*)C;
#pragma unroll
        for (int j = 0; j < 4; ++j)
#pragma unroll
            for (int r = 0; r < 8; ++r)
                Cf[(size_t)(mbase + r) * ldc + n0 + j * 16 + nn] = acc[j][r];
    }
}

// ---------------- flash attention (GQA), bf16 WMMA ----------------
// Grid: (S/64, HEADS, BATCH). Block = 128 threads (4 waves, 16 Q rows each).
// Softmax scale pre-folded into Q. V consumed from global V^T (b,hk,d,s).
#define KSTR 136   // K tile [kv][d] stride (272B rows)
#define VSTR 72    // V^T tile [d][kv] stride (144B rows)
#define PSTR 72    // P staging stride per wave

__global__ __launch_bounds__(128)
void flash_gqa_kernel(const unsigned short* __restrict__ Qb,
                      const unsigned short* __restrict__ Kb,
                      const unsigned short* __restrict__ VTb,
                      unsigned short* __restrict__ Yb) {
    __shared__ unsigned short lK[64 * KSTR];
    __shared__ unsigned short lV[HD * VSTR];
    __shared__ unsigned short lP[4 * 16 * PSTR];

    const int tid  = threadIdx.x;
    const int lane = tid & 31;
    const int wave = tid >> 5;
    const int qt = blockIdx.x;
    const int h  = blockIdx.y;
    const int b  = blockIdx.z;
    const int hk = h / GROUPS;
    const int q0 = qt * 64 + wave * 16;

    // --- hoisted tile-load geometry ---
    // K tile: 8 x 16B chunks/thread: kv = tid>>4 (+8i), d8 = (tid&15)*8
    const unsigned short* pK[8];
    int kslot[8];
    {
        const int kvb = tid >> 4;
        const int d8  = (tid & 15) * 8;
        const unsigned short* kbase =
            Kb + (size_t)b * SEQ * KVHEADS * HD + (size_t)hk * HD + d8;
#pragma unroll
        for (int i = 0; i < 8; ++i) {
            int kv = kvb + 8 * i;
            pK[i]    = kbase + (size_t)kv * KVHEADS * HD;
            kslot[i] = kv * KSTR + d8;
        }
    }
    // V^T tile: 8 x 16B chunks/thread: d = tid>>3 (+16i), kv8 = (tid&7)*8
    const unsigned short* pV[8];
    int vslot[8];
    {
        const int db  = tid >> 3;
        const int kv8 = (tid & 7) * 8;
        const unsigned short* vbase =
            VTb + ((size_t)b * KVHEADS + hk) * HD * SEQ + kv8;
#pragma unroll
        for (int i = 0; i < 8; ++i) {
            int d = db + 16 * i;
            pV[i]    = vbase + (size_t)d * SEQ;
            vslot[i] = d * VSTR + kv8;
        }
    }

    // Preload Q (already scaled by 1/sqrt(HD)) as 4 A-fragments
    v16bf qf[4];
    {
        const int mrow  = q0 + (lane & 15);
        const size_t qrow = (((size_t)b * SEQ + mrow) * HEADS + h) * HD;
        const int khalf = (lane >> 4) * 8;
#pragma unroll
        for (int ks = 0; ks < 4; ++ks)
            qf[ks] = load_frag2(Qb + qrow + ks * 32 + khalf,
                                Qb + qrow + ks * 32 + khalf + 16);
    }

    float m_s[8], l_s[8];
    v8f o[8] = {};
#pragma unroll
    for (int r = 0; r < 8; ++r) { m_s[r] = -3.0e38f; l_s[r] = 0.0f; }

    const size_t kAdv = (size_t)64 * KVHEADS * HD;   // per-tile K advance
    for (int it = 0; it < SEQ / 64; ++it) {
        __syncthreads();
#pragma unroll
        for (int i = 0; i < 8; ++i)
            *(v8us*)(&lK[kslot[i]]) = *(const v8us*)(pK[i] + (size_t)it * kAdv);
#pragma unroll
        for (int i = 0; i < 8; ++i)
            *(v8us*)(&lV[vslot[i]]) = *(const v8us*)(pV[i] + (size_t)it * 64);
        __syncthreads();

        // S = Q * K^T  (16 q-rows x 64 kv-cols per wave)
        v8f sacc[4] = {};
#pragma unroll
        for (int ks = 0; ks < 4; ++ks) {
            const int dh = ks * 32 + (lane >> 4) * 8;
            v16bf bfr[4];
#pragma unroll
            for (int j = 0; j < 4; ++j) {
                const unsigned short* p = &lK[(j * 16 + (lane & 15)) * KSTR + dh];
                bfr[j] = load_frag2(p, p + 16);
            }
#pragma unroll
            for (int j = 0; j < 4; ++j)
                sacc[j] = __builtin_amdgcn_wmma_f32_16x16x32_bf16(
                    false, qf[ks], false, bfr[j], (short)0, sacc[j], false, false);
        }

        // Online softmax; row r lives in VGPR slot r across each 16-lane half.
        float mnew[8], alpha[8];
#pragma unroll
        for (int r = 0; r < 8; ++r) {
            float t = fmaxf(fmaxf(sacc[0][r], sacc[1][r]),
                            fmaxf(sacc[2][r], sacc[3][r]));
#pragma unroll
            for (int off = 8; off >= 1; off >>= 1)
                t = fmaxf(t, __shfl_xor(t, off, 32));
            mnew[r]  = fmaxf(m_s[r], t);
            alpha[r] = __expf(m_s[r] - mnew[r]);
            m_s[r]   = mnew[r];
        }
#pragma unroll
        for (int r = 0; r < 8; ++r) {
            float s = 0.0f;
#pragma unroll
            for (int j = 0; j < 4; ++j) {
                float p = __expf(sacc[j][r] - mnew[r]);
                sacc[j][r] = p;
                s += p;
            }
#pragma unroll
            for (int off = 8; off >= 1; off >>= 1)
                s += __shfl_xor(s, off, 32);
            l_s[r] = l_s[r] * alpha[r] + s;
        }
#pragma unroll
        for (int dj = 0; dj < 8; ++dj)
#pragma unroll
            for (int r = 0; r < 8; ++r) o[dj][r] *= alpha[r];

        // Re-shape P: C-layout -> LDS -> A-fragment layout (per-wave region)
        unsigned short* myP = &lP[wave * 16 * PSTR];
#pragma unroll
        for (int j = 0; j < 4; ++j)
#pragma unroll
            for (int r = 0; r < 8; ++r) {
                int row = r + ((lane >> 4) * 8);
                myP[row * PSTR + j * 16 + (lane & 15)] = f32_to_bf16(sacc[j][r]);
            }
        // Same-wave LDS RAW: drain DS counter before fragment reload
        asm volatile("s_wait_dscnt 0" ::: "memory");

        // O += P * V  (two 32-wide kv K-steps)
#pragma unroll
        for (int kstep = 0; kstep < 2; ++kstep) {
            const unsigned short* pp =
                &myP[(lane & 15) * PSTR + kstep * 32 + (lane >> 4) * 8];
            v16bf pf = load_frag2(pp, pp + 16);
            const int kvh = kstep * 32 + (lane >> 4) * 8;
            v16bf vf[8];
#pragma unroll
            for (int dj = 0; dj < 8; ++dj) {
                const unsigned short* p = &lV[(dj * 16 + (lane & 15)) * VSTR + kvh];
                vf[dj] = load_frag2(p, p + 16);
            }
#pragma unroll
            for (int dj = 0; dj < 8; ++dj)
                o[dj] = __builtin_amdgcn_wmma_f32_16x16x32_bf16(
                    false, pf, false, vf[dj], (short)0, o[dj], false, false);
        }
    }

    // Normalize and store y (bf16, layout (b, s, h*HD+d))
    float inv[8];
#pragma unroll
    for (int r = 0; r < 8; ++r) inv[r] = 1.0f / l_s[r];
#pragma unroll
    for (int dj = 0; dj < 8; ++dj)
#pragma unroll
        for (int r = 0; r < 8; ++r) {
            int row = q0 + r + ((lane >> 4) * 8);
            Yb[(((size_t)b * SEQ + row) * HEADS + h) * HD + dj * 16 + (lane & 15)] =
                f32_to_bf16(o[dj][r] * inv[r]);
        }
}

// ---------------- host-side orchestration ----------------
extern "C" void kernel_launch(void* const* d_in, const int* in_sizes, int n_in,
                              void* d_out, int out_size, void* d_ws, size_t ws_size,
                              hipStream_t stream) {
    (void)in_sizes; (void)n_in; (void)out_size; (void)ws_size;
    const float* x  = (const float*)d_in[0];
    const float* wq = (const float*)d_in[1];
    const float* wk = (const float*)d_in[2];
    const float* wv = (const float*)d_in[3];
    const float* wo = (const float*)d_in[4];
    float* out = (float*)d_out;

    // workspace layout (bytes)
    char* ws = (char*)d_ws;
    constexpr size_t SZ_XBF  = (size_t)BS * EMBED * 2;             // 16 MiB
    constexpr size_t SZ_WQBF = (size_t)EMBED * EMBED * 2;          // 32 MiB
    constexpr size_t SZ_WKBF = (size_t)EMBED * (KVHEADS * HD) * 2; //  8 MiB
    constexpr size_t SZ_QF   = (size_t)BS * EMBED * 4;             // 32 MiB
    constexpr size_t SZ_KF   = (size_t)BS * (KVHEADS * HD) * 4;    //  8 MiB
    constexpr size_t SZ_QBF  = (size_t)BS * EMBED * 2;             // 16 MiB
    constexpr size_t SZ_KBF  = (size_t)BS * (KVHEADS * HD) * 2;    //  4 MiB

    size_t off = 0;
    unsigned short* x_bf  = (unsigned short*)(ws + off); off += SZ_XBF;
    unsigned short* wq_bf = (unsigned short*)(ws + off); off += SZ_WQBF;
    unsigned short* wk_bf = (unsigned short*)(ws + off); off += SZ_WKBF;
    unsigned short* wv_bf = (unsigned short*)(ws + off); off += SZ_WKBF;
    unsigned short* wo_bf = (unsigned short*)(ws + off); off += SZ_WQBF;
    float*          q_f   = (float*)(ws + off);          off += SZ_QF;
    float*          k_f   = (float*)(ws + off);          off += SZ_KF;
    unsigned short* q_bf  = (unsigned short*)(ws + off); off += SZ_QBF;
    unsigned short* k_bf  = (unsigned short*)(ws + off); off += SZ_KBF;
    unsigned short* v_bf  = (unsigned short*)(ws + off); off += SZ_KBF;
    unsigned short* vT_bf = (unsigned short*)(ws + off); off += SZ_KBF;
    unsigned short* y_bf  = (unsigned short*)(ws + off); off += SZ_QBF;

    // 1) f32 -> bf16 conversions
    cvt_bf16_kernel<<<4096, 256, 0, stream>>>(x,  x_bf,  (size_t)BS * EMBED);
    cvt_bf16_kernel<<<4096, 256, 0, stream>>>(wq, wq_bf, (size_t)EMBED * EMBED);
    cvt_bf16_kernel<<<4096, 256, 0, stream>>>(wk, wk_bf, (size_t)EMBED * KVHEADS * HD);
    cvt_bf16_kernel<<<4096, 256, 0, stream>>>(wv, wv_bf, (size_t)EMBED * KVHEADS * HD);
    cvt_bf16_kernel<<<4096, 256, 0, stream>>>(wo, wo_bf, (size_t)EMBED * EMBED);

    // 2) Q/K/V projections (WMMA bf16, f32 accumulate)
    gemm_bf16_wmma<<<dim3(EMBED / BN, BS / BM), 256, 0, stream>>>(
        x_bf, wq_bf, (void*)q_f, BS, EMBED, EMBED, EMBED, 0);
    gemm_bf16_wmma<<<dim3((KVHEADS * HD) / BN, BS / BM), 256, 0, stream>>>(
        x_bf, wk_bf, (void*)k_f, BS, KVHEADS * HD, EMBED, KVHEADS * HD, 0);
    gemm_bf16_wmma<<<dim3((KVHEADS * HD) / BN, BS / BM), 256, 0, stream>>>(
        x_bf, wv_bf, (void*)v_bf, BS, KVHEADS * HD, EMBED, KVHEADS * HD, 1);

    // 3) RoPE (f32 in, bf16 out); softmax scale folded into Q
    {
        const float qscale = 0.08838834764831845f;  // 1/sqrt(128)
        int tq = BATCH * SEQ * HEADS * (HD / 2);
        int tk = BATCH * SEQ * KVHEADS * (HD / 2);
        rope_kernel<<<(tq + 255) / 256, 256, 0, stream>>>(q_f, q_bf, HEADS, qscale);
        rope_kernel<<<(tk + 255) / 256, 256, 0, stream>>>(k_f, k_bf, KVHEADS, 1.0f);
    }

    // 3b) V -> V^T (b,hk,d,s) so flash can load it vectorized
    {
        int tv = BATCH * SEQ * KVHEADS * HD;
        transpose_v_kernel<<<(tv + 255) / 256, 256, 0, stream>>>(v_bf, vT_bf);
    }

    // 4) flash attention
    flash_gqa_kernel<<<dim3(SEQ / 64, HEADS, BATCH), 128, 0, stream>>>(
        q_bf, k_bf, vT_bf, y_bf);

    // 5) output projection -> f32 d_out
    gemm_bf16_wmma<<<dim3(EMBED / BN, BS / BM), 256, 0, stream>>>(
        y_bf, wo_bf, (void*)out, BS, EMBED, EMBED, EMBED, 0);
}
